// ModulatedConv_23278722744487
// MI455X (gfx1250) — matrix operand order repaired and checked by reading, not compiled, not run on hardware
//
#include <hip/hip_runtime.h>
#include <hip/hip_bf16.h>

typedef __attribute__((ext_vector_type(16))) _Float16 v16h;
typedef __attribute__((ext_vector_type(8)))  float    v8f;

#define BB   8
#define CC   64
#define HH   96
#define WW   96
#define OO   64
#define KS   3
#define KKQ  9           // K*K
#define HO   96
#define WO   96
#define KDIM (CC*KKQ)    // 576
#define MT   32          // positions per block (M tile)
#define NWAVES 8
#define TPB  (NWAVES*32)

#define SP_BYTES (KKQ*MT*8*4)     // 9216 : bilinear params
#define SA_BYTES (2*MT*32*2)      // 4096 : A double buffer
#define SW_BYTES (OO*KDIM*2)      // 73728: f16 weights (fallback path only)

union V16HU { v16h h; uint4 u[2]; };

// ---- one-shot: transpose weights (o,c,kk) -> (o, kk*64+c), f32 -> f16, into d_ws ----
__global__ __launch_bounds__(256) void mdcn_prep_w(const float* __restrict__ wgt,
                                                   _Float16* __restrict__ wT) {
  int i = blockIdx.x * 256 + threadIdx.x;
  if (i < OO*KDIM) {
    int o  = i / KDIM;
    int r  = i - o*KDIM;
    int kk = r >> 6;
    int c  = r & 63;
    wT[i] = (_Float16)wgt[(o*CC + c)*KKQ + kk];
  }
}

template<bool WGLOBAL>
__global__ __launch_bounds__(TPB) void mdcn_wmma_kernel(
    const float* __restrict__ x, const float* __restrict__ off,
    const float* __restrict__ msk, const _Float16* __restrict__ wT,
    const float* __restrict__ wgt, const float* __restrict__ bias,
    float* __restrict__ out)
{
  extern __shared__ char smem[];
  float*    sP = (float*)smem;                      // [KKQ][MT][8] {w00..w11, at, pad}
  _Float16* sA = (_Float16*)(smem + SP_BYTES);      // [2][MT][32] double-buffered
  _Float16* sW = (_Float16*)(smem + SP_BYTES + SA_BYTES);  // fallback only

  const int tid = threadIdx.x;

  if (!WGLOBAL) {
    // stage + transpose weights into LDS (f32 -> f16)
    for (int i = tid; i < OO*KDIM; i += TPB) {
      int o  = i / KDIM;
      int r  = i - o*KDIM;
      int kk = r >> 6;
      int c  = r & 63;
      sW[i] = (_Float16)wgt[(o*CC + c)*KKQ + kk];
    }
  }

  // ---- this block's 32 output positions: fixed (b, ho), wo = w0..w0+31 ----
  const int p0 = blockIdx.x * MT;
  const int b  = p0 / (HO*WO);
  const int rp = p0 - b*(HO*WO);
  const int ho = rp / WO;
  const int w0 = rp - ho*WO;

  // ---- precompute remapped bilinear params for all (kk, m) once ----
  // Separable remap: base corner at = by*W+bx with bx=clamp(x0,0,W-2), by=clamp(y0,0,H-2);
  // border validity folded into the 4 weights (exact, incl. x0==-1 / x0==W-1 cases).
  for (int q = tid; q < KKQ*MT; q += TPB) {
    const int mq  = q & (MT-1);
    const int kk  = q >> 5;
    const int woq = w0 + mq;
    const int ky = kk / KS, kx = kk - ky*KS;
    const float offy = off[(((b*2*KKQ) + 2*kk + 0)*HO + ho)*WO + woq];
    const float offx = off[(((b*2*KKQ) + 2*kk + 1)*HO + ho)*WO + woq];
    const float mval = msk[((b*KKQ + kk)*HO + ho)*WO + woq];
    const float py = (float)(ho - 1 + ky) + offy;
    const float px = (float)(woq - 1 + kx) + offx;
    const float y0f = floorf(py), x0f = floorf(px);
    const float ly = py - y0f,  lx = px - x0f;
    const int y0 = (int)y0f, x0i = (int)x0f;

    float wl = 0.f, wr = 0.f;
    if (x0i >= 0 && x0i <= WW-2)      { wl = 1.f - lx; wr = lx; }
    else if (x0i == -1)               { wl = lx; }            // only x1=0 valid -> bx=0
    else if (x0i == WW-1)             { wr = 1.f - lx; }      // only x0 valid -> bx+1
    float wt = 0.f, wb = 0.f;
    if (y0 >= 0 && y0 <= HH-2)        { wt = 1.f - ly; wb = ly; }
    else if (y0 == -1)                { wt = ly; }
    else if (y0 == HH-1)              { wb = 1.f - ly; }

    const int bx = min(max(x0i, 0), WW-2);
    const int by = min(max(y0,  0), HH-2);

    float* p = sP + q*8;
    p[0] = wt*wl*mval;  // weight for x[at]
    p[1] = wt*wr*mval;  // weight for x[at+1]
    p[2] = wb*wl*mval;  // weight for x[at+W]
    p[3] = wb*wr*mval;  // weight for x[at+W+1]
    ((int*)p)[4] = by*WW + bx;
  }

  const int m  = tid & (MT-1);     // position this thread fills in A
  const int cg = tid >> 5;         // channel group 0..7 (4 channels each)

  const float* xb = x + (size_t)b*CC*HH*WW;

  const int wave  = tid >> 5;
  const int msub  = wave >> 2;     // 0..1 : M subtile
  const int nsl   = wave & 3;      // 0..3 : N slice
  const int lane  = tid & 31;
  const int nrow  = lane & 15;
  const int khalf = lane >> 4;

  // B rows: stream from global (L2-resident 72KB) or from LDS (fallback)
  const _Float16* wrow = WGLOBAL ? (wT + (size_t)(nsl*16 + nrow)*KDIM)
                                 : (sW + (size_t)(nsl*16 + nrow)*KDIM);

  v8f acc = {};

  __syncthreads();

  for (int kk = 0; kk < KKQ; ++kk) {
    // params for (kk, m): b128 (weights) + b32 (base address)
    const float* p = sP + (kk*MT + m)*8;
    const float4 pw = *(const float4*)p;
    const int    at = ((const int*)p)[4];
    const float* xq = xb + at;

#pragma unroll
    for (int chalf = 0; chalf < 2; ++chalf) {
      // ---- fill A chunk: 32 positions x 32 channels (fixed kk) ----
      _Float16* sAc = sA + chalf * (MT*32);
      const int cbase = chalf*32 + cg*4;
#pragma unroll
      for (int j = 0; j < 4; ++j) {
        const float* xc = xq + (size_t)(cbase + j)*HH*WW;
        // two adjacent-pair loads -> global_load_b64 (+imm 384 for bottom row)
        float v = pw.x*xc[0] + pw.y*xc[1] + pw.z*xc[WW] + pw.w*xc[WW+1];
        sAc[m*32 + cg*4 + j] = (_Float16)v;
      }
      __syncthreads();

      // ---- fragments per ISA 7.12.2 layouts ----
      V16HU af, bf;
      // A 16x32 f16: lanes 0-15 -> K {0..7,16..23}; lanes 16-31 -> K {8..15,24..31}
      const uint4* paf = (const uint4*)(sAc + (msub*16 + nrow)*32 + khalf*8);
      af.u[0] = paf[0];
      af.u[1] = paf[2];
      // B 32x16 f16: lanes 0-15 -> K {0..7,16..23}; lanes 16-31 -> K {8..15,24..31}; lane&15 = N (output channel)
      const int kb = kk*64 + chalf*32 + khalf*8;
      const uint4* pb = (const uint4*)(wrow + kb);
      bf.u[0] = pb[0];
      bf.u[1] = pb[2];

      acc = __builtin_amdgcn_wmma_f32_16x16x32_f16(false, af.h, false, bf.h,
                                                   (short)0, acc, false, false);
    }
  }

  // ---- epilogue: D 16x16 f32: lane&15 = N (channel), VGPR r -> M = r + 8*(lane>>4) ----
  const int o  = nsl*16 + nrow;
  const float bo = bias[o];
  float* outp = out + (((size_t)b*OO + o)*HO + ho)*WO + w0 + msub*16;
#pragma unroll
  for (int r = 0; r < 8; ++r) {
    outp[r + 8*khalf] = acc[r] + bo;
  }
}

extern "C" void kernel_launch(void* const* d_in, const int* in_sizes, int n_in,
                              void* d_out, int out_size, void* d_ws, size_t ws_size,
                              hipStream_t stream) {
  const float* x    = (const float*)d_in[0];
  const float* off  = (const float*)d_in[1];
  const float* msk  = (const float*)d_in[2];
  const float* wgt  = (const float*)d_in[3];
  const float* bias = (const float*)d_in[4];
  float* out = (float*)d_out;

  const int nblocks = (BB*HO*WO) / MT;                 // 2304

  if (ws_size >= (size_t)SW_BYTES) {
    // one-shot f16 weight transpose into workspace, then high-occupancy main kernel
    mdcn_prep_w<<<dim3((OO*KDIM + 255)/256), dim3(256), 0, stream>>>(
        wgt, (_Float16*)d_ws);
    mdcn_wmma_kernel<true><<<dim3(nblocks), dim3(TPB),
                             SP_BYTES + SA_BYTES, stream>>>(
        x, off, msk, (const _Float16*)d_ws, wgt, bias, out);
  } else {
    mdcn_wmma_kernel<false><<<dim3(nblocks), dim3(TPB),
                              SP_BYTES + SA_BYTES + SW_BYTES, stream>>>(
        x, off, msk, (const _Float16*)nullptr, wgt, bias, out);
  }
}
